// CLPredictor_64476049047811
// MI455X (gfx1250) — matrix operand order repaired and checked by reading, not compiled, not run on hardware
//
#include <hip/hip_runtime.h>
#include <hip/hip_bf16.h>

// ---------------------------------------------------------------------------
// CLPredictor for MI455X (gfx1250), wave32 + WMMA bf16 16x16x32.
//
// Shapes: B=2, N=256, D=512, H=1024, C=6.
// Dominant op: [B*N*N, H] @ [H, H/2]  (137 GFLOP) -> fused bf16 WMMA GEMM,
// hq built on the fly in LDS from a_i + b_j + bs1 (never materialized).
// All weights pre-transposed to [N][K] bf16 so LDS staging is pure b128.
// ---------------------------------------------------------------------------

typedef unsigned short u16;
typedef __attribute__((ext_vector_type(16))) __bf16 v16bf;
typedef __attribute__((ext_vector_type(8)))  float  v8f;
typedef __attribute__((ext_vector_type(4)))  unsigned int v4u;

union FragBF { v16bf v; v4u q[2]; };
union Pack8  { v4u q; u16 s[8]; };

__device__ __forceinline__ u16 f2bf(float f) {
    unsigned u = __float_as_uint(f);
    unsigned r = u + 0x7fffu + ((u >> 16) & 1u);   // round-to-nearest-even
    return (u16)(r >> 16);
}

// ------------------------------ fp32 -> bf16 -------------------------------
__global__ void k_cvt_bf16(const float* __restrict__ in, u16* __restrict__ out, int n) {
    int i = blockIdx.x * 256 + threadIdx.x;
    if (i < n) out[i] = f2bf(in[i]);
}

// ------------------ fp32 [K][N] -> bf16 transposed [N][K] ------------------
__global__ void k_cvt_t_bf16(const float* __restrict__ in, u16* __restrict__ out,
                             int K, int N) {
    int idx = blockIdx.x * 256 + threadIdx.x;            // idx = n*K + k
    if (idx >= K * N) return;
    int n = idx / K, k = idx - n * K;
    out[idx] = f2bf(in[(size_t)k * N + n]);
}

// ------------------------------ zero init ----------------------------------
__global__ void k_zero(float* __restrict__ p, int n) {
    int i = blockIdx.x * 256 + threadIdx.x;
    if (i < n) p[i] = 0.0f;
}

// ---------------------------------------------------------------------------
// Generic bf16 WMMA GEMM:  out[M,N] = act(A[M,K] @ W[K,N] + bias)
// W supplied PRE-TRANSPOSED as WT[N][K] bf16.
// WG = 256 threads (8 waves), tile M=64 x N=128, K-step 32.
// Wave w: M-block (w&3), N-blocks (w>>2)*4 .. +3  (16x16 tiles, v8f accum).
// LDS: a_s[64][40] bf16, wT_s[128][40] bf16 -- all staging is b128.
// ---------------------------------------------------------------------------
__global__ __launch_bounds__(256) void k_gemm_bf16(
    const u16* __restrict__ A, const u16* __restrict__ WT,
    const float* __restrict__ bias, float* __restrict__ outF, u16* __restrict__ outB,
    int K, int N, int doRelu)
{
    extern __shared__ char smem[];
    u16* a_s  = (u16*)smem;                 // [64][40]
    u16* wT_s = (u16*)(smem + 64 * 40 * 2); // [128][40]

    const int t    = threadIdx.x;
    const int lane = t & 31;
    const int w    = t >> 5;
    const int r    = lane & 15;
    const int hi   = lane >> 4;
    const int m0   = blockIdx.x * 64;
    const int n0   = blockIdx.y * 128;
    const int mb   = w & 3;
    const int ng   = (w >> 2) * 4;

    v8f acc[4];
    for (int q = 0; q < 4; q++)
        for (int e = 0; e < 8; e++) acc[q][e] = 0.0f;

    const int sm  = t >> 2, sk0 = (t & 3) * 8;   // A staging: 64 rows x 4 thr x 8
    const int wn  = t >> 1, wk0 = (t & 1) * 16;  // WT staging: 128 rows x 2 thr x 16

    for (int K0 = 0; K0 < K; K0 += 32) {
        // stage A tile [64][32] (b128 in, b128 out)
        v4u av = *(const v4u*)(A + (size_t)(m0 + sm) * K + K0 + sk0);
        *(v4u*)(a_s + sm * 40 + sk0) = av;
        // stage WT k-slice: wT_s[n][k] = WT[n0+n][K0+k]  (pure b128, no scatter)
        const u16* wp = WT + (size_t)(n0 + wn) * K + K0 + wk0;
        v4u w0 = *(const v4u*)(wp);
        v4u w1 = *(const v4u*)(wp + 8);
        *(v4u*)(wT_s + wn * 40 + wk0)     = w0;
        *(v4u*)(wT_s + wn * 40 + wk0 + 8) = w1;
        if (K0 + 32 < K) __builtin_prefetch(wp + 32, 0, 1);   // next k-slice (L2)
        __syncthreads();

        // A fragment (16x32): elem e -> K = hi*16 + e, M = r
        FragBF af;
        const u16* ap = a_s + (mb * 16 + r) * 40;
        af.q[0] = *(const v4u*)(ap + hi * 16);
        af.q[1] = *(const v4u*)(ap + hi * 16 + 8);

        // B fragments (32x16): elem e -> K = hi*16 + e, N = r  (double-buffered)
        FragBF bcur, bnxt;
        const u16* bp0 = wT_s + (ng * 16 + r) * 40 + hi * 16;
        bcur.q[0] = *(const v4u*)(bp0);
        bcur.q[1] = *(const v4u*)(bp0 + 8);
        for (int nb = 0; nb < 4; nb++) {
            if (nb < 3) {
                const u16* bp = wT_s + ((ng + nb + 1) * 16 + r) * 40 + hi * 16;
                bnxt.q[0] = *(const v4u*)(bp);
                bnxt.q[1] = *(const v4u*)(bp + 8);
            }
            acc[nb] = __builtin_amdgcn_wmma_f32_16x16x32_bf16(
                false, af.v, false, bcur.v, (short)0, acc[nb], false, false);
            bcur = bnxt;
        }
        __syncthreads();
    }

    // epilogue: D elem e -> row = e + hi*8, col = r
    for (int nb = 0; nb < 4; nb++) {
        int ncol = n0 + (ng + nb) * 16 + r;
        float bv = bias ? bias[ncol] : 0.0f;
        for (int e = 0; e < 8; e++) {
            int row = m0 + mb * 16 + hi * 8 + e;
            float v = acc[nb][e] + bv;
            if (doRelu) v = fmaxf(v, 0.0f);
            size_t idx = (size_t)row * N + ncol;
            if (outF) outF[idx] = v;
            if (outB) outB[idx] = f2bf(v);
        }
    }
}

// ---------------------------------------------------------------------------
// Per-segment head: logit_p = h2 @ W3 + b3, log-softmax, hat_p, seg_y, loss1.
// rows = B*N = 512, one thread per row.
// ---------------------------------------------------------------------------
__global__ void k_head(const float* __restrict__ h2f, const float* __restrict__ W3,
                       const float* __restrict__ b3, const long long* __restrict__ y,
                       float* __restrict__ dout, int* __restrict__ segy,
                       float* __restrict__ lossAcc)
{
    int rrow = blockIdx.x * 256 + threadIdx.x;
    if (rrow >= 512) return;
    float lg[6];
    for (int c = 0; c < 6; c++) lg[c] = b3[c];
    const float* hr = h2f + (size_t)rrow * 512;
    for (int k = 0; k < 512; k++) {
        float h = hr[k];
        for (int c = 0; c < 6; c++) lg[c] = fmaf(h, W3[k * 6 + c], lg[c]);
    }
    float mx = lg[0];
    for (int c = 1; c < 6; c++) mx = fmaxf(mx, lg[c]);
    float s = 0.0f;
    for (int c = 0; c < 6; c++) s += __expf(lg[c] - mx);
    float lse = mx + __logf(s);

    int seg = 0; long long best = y[(size_t)rrow * 6];
    for (int c = 1; c < 6; c++) {
        long long v = y[(size_t)rrow * 6 + c];
        if (v > best) { best = v; seg = c; }
    }
    for (int c = 0; c < 6; c++)
        dout[1 + (size_t)rrow * 6 + c] = __expf(lg[c] - lse);     // hat_p
    segy[rrow] = seg;
    dout[1 + 3072 + 3072 + rrow] = (float)seg;                    // seg_y
    atomicAdd(&lossAcc[0], -(lg[seg] - lse) * (1.0f / 512.0f));   // loss1
}

// ---------------------------------------------------------------------------
// Fused pairwise kernel. One WG = (batch b, row i, 64 consecutive j's).
// Phase 1: z[64][512] = relu( relu(a_i + b_j + bs1) @ Ws2 + bs2 )  via WMMA.
//   K loop (32/step): hq k-slice built in LDS (fp32 add+relu -> packed bf16,
//   single b128 store/thread); Ws2T k-slice staged with pure b128 traffic;
//   8 waves x 16 tiles (4 Mblk x 32 Nblk), B frags double-buffered.
// Phase 2: logit_q = z @ Ws3 + bs3, 2-way log-softmax, loss2, tilde accum.
// Dynamic LDS: max(staging 46KB, z 132KB) = 132096 B (time-disjoint overlap).
// ---------------------------------------------------------------------------
__global__ __launch_bounds__(256) void k_pair(
    const float* __restrict__ a_f, const float* __restrict__ b_f,
    const float* __restrict__ bs1, const u16* __restrict__ Ws2T,
    const float* __restrict__ bs2, const float* __restrict__ Ws3,
    const float* __restrict__ bs3, const float* __restrict__ hatp,
    const int* __restrict__ segy, float* __restrict__ tildeAcc,
    float* __restrict__ lossAcc)
{
    extern __shared__ char smem[];
    u16*   hq_s = (u16*)smem;                 // [64][40]   (phase 1)
    u16*   wT_s = (u16*)(smem + 64 * 40 * 2); // [512][40]  (phase 1)
    float* z_s  = (float*)smem;               // [64][516]  (phase 1 epilogue ->2)
    __shared__ float red_s[64 * 6];
    __shared__ float l2_s[64];

    const int t    = threadIdx.x;
    const int lane = t & 31;
    const int w    = t >> 5;
    const int r    = lane & 15;
    const int hi   = lane >> 4;

    const int wg = blockIdx.x;
    const int b  = wg >> 10;          // 1024 WGs per batch
    const int i  = (wg >> 2) & 255;
    const int j0 = (wg & 3) * 64;

    const float* arow  = a_f + (size_t)(b * 256 + i) * 1024;
    const float* brows = b_f + (size_t)(b * 256 + j0) * 1024;

    const int mb    = w & 3;
    const int nhalf = w >> 2;

    v8f acc[16];
    for (int q = 0; q < 16; q++)
        for (int e = 0; e < 8; e++) acc[q][e] = 0.0f;

    const int sm = t >> 2, sk0 = (t & 3) * 8;   // hq staging: 64 rows x 4 thr x 8

    for (int K0 = 0; K0 < 1024; K0 += 32) {
        // build hq k-slice: relu(a[i][k] + b[j0+m][k] + bs1[k]) -> packed bf16
        {
            Pack8 pk;
            const float* ar = arow + K0 + sk0;
            const float* br = brows + (size_t)sm * 1024 + K0 + sk0;
            const float* sr = bs1 + K0 + sk0;
            for (int u = 0; u < 8; u++)
                pk.s[u] = f2bf(fmaxf(ar[u] + br[u] + sr[u], 0.0f));
            *(v4u*)(hq_s + sm * 40 + sk0) = pk.q;
        }
        // stage Ws2T k-slice: wT_s[n][k] = Ws2T[n][K0+k]  (2 rows per thread)
        for (int q2 = 0; q2 < 2; q2++) {
            int n = t + q2 * 256;
            const u16* wp = Ws2T + (size_t)n * 1024 + K0;
            v4u x0 = *(const v4u*)(wp);
            v4u x1 = *(const v4u*)(wp + 8);
            v4u x2 = *(const v4u*)(wp + 16);
            v4u x3 = *(const v4u*)(wp + 24);
            u16* d = wT_s + n * 40;
            *(v4u*)(d)      = x0;
            *(v4u*)(d + 8)  = x1;
            *(v4u*)(d + 16) = x2;
            *(v4u*)(d + 24) = x3;
            if (K0 + 32 < 1024) __builtin_prefetch(wp + 32, 0, 1);  // next slice
        }
        __syncthreads();

        // A fragment from hq
        FragBF af;
        const u16* ap = hq_s + (mb * 16 + r) * 40;
        af.q[0] = *(const v4u*)(ap + hi * 16);
        af.q[1] = *(const v4u*)(ap + hi * 16 + 8);

        // 16 B fragments, double-buffered so WMMAs overlap the ds loads
        FragBF bcur, bnxt;
        const u16* bp0 = wT_s + (nhalf * 256 + r) * 40 + hi * 16;
        bcur.q[0] = *(const v4u*)(bp0);
        bcur.q[1] = *(const v4u*)(bp0 + 8);
        for (int nt = 0; nt < 16; nt++) {
            if (nt < 15) {
                const u16* bp = wT_s + (nhalf * 256 + (nt + 1) * 16 + r) * 40 + hi * 16;
                bnxt.q[0] = *(const v4u*)(bp);
                bnxt.q[1] = *(const v4u*)(bp + 8);
            }
            acc[nt] = __builtin_amdgcn_wmma_f32_16x16x32_bf16(
                false, af.v, false, bcur.v, (short)0, acc[nt], false, false);
            bcur = bnxt;
        }
        __syncthreads();
    }

    // z = relu(acc + bs2) -> LDS (overwrites staging; all WMMA done)
    for (int nt = 0; nt < 16; nt++) {
        int ncol = nhalf * 256 + nt * 16 + r;
        float bv = bs2[ncol];
        for (int e = 0; e < 8; e++) {
            int mloc = mb * 16 + hi * 8 + e;
            z_s[mloc * 516 + ncol] = fmaxf(acc[nt][e] + bv, 0.0f);
        }
    }
    __syncthreads();

    // Phase 2: head over the 64 pairs of this WG (shared row i)
    if (t < 64) {
        int j = j0 + t;
        float l0 = bs3[0], l1 = bs3[1];
        const float* zr = z_s + t * 516;
        for (int c = 0; c < 512; c++) {
            float zv = zr[c];
            l0 = fmaf(zv, Ws3[c * 2 + 0], l0);
            l1 = fmaf(zv, Ws3[c * 2 + 1], l1);
        }
        float mx  = fmaxf(l0, l1);
        float lse = mx + __logf(__expf(l0 - mx) + __expf(l1 - mx));
        float lq0 = l0 - lse, lq1 = l1 - lse;
        int same = (segy[b * 256 + i] == segy[b * 256 + j]) ? 1 : 0;
        l2_s[t] = -(same ? lq1 : lq0) * (1.0f / 131072.0f);       // loss2 part
        float hq1 = __expf(lq1);                                  // hat_q_1
        const float* hp = hatp + (size_t)(b * 256 + j) * 6;
        for (int c = 0; c < 6; c++) red_s[t * 6 + c] = hq1 * hp[c];
    }
    __syncthreads();
    if (t < 6) {
        float s = 0.0f;
        for (int m = 0; m < 64; m++) s += red_s[m * 6 + t];
        atomicAdd(&tildeAcc[(size_t)(b * 256 + i) * 6 + t], s);
    }
    if (t == 0) {
        float s = 0.0f;
        for (int m = 0; m < 64; m++) s += l2_s[m];
        atomicAdd(&lossAcc[1], s);
    }
}

// --------------------------- finalize outputs ------------------------------
__global__ void k_final(const float* __restrict__ tildeAcc,
                        const float* __restrict__ lossAcc, float* __restrict__ dout)
{
    int rrow = blockIdx.x * 256 + threadIdx.x;
    if (rrow < 512) {
        float v[6], s = 0.0f;
        for (int c = 0; c < 6; c++) { v[c] = tildeAcc[rrow * 6 + c]; s += v[c]; }
        float inv = 1.0f / s;
        for (int c = 0; c < 6; c++) dout[1 + 3072 + rrow * 6 + c] = v[c] * inv;
    }
    if (rrow == 0) dout[0] = lossAcc[0] + lossAcc[1];
}

// ---------------------------------------------------------------------------
extern "C" void kernel_launch(void* const* d_in, const int* in_sizes, int n_in,
                              void* d_out, int out_size, void* d_ws, size_t ws_size,
                              hipStream_t stream) {
    const float*     agg = (const float*)d_in[0];      // [2,256,512]
    const long long* y   = (const long long*)d_in[1];  // [2,256,6] int64 one-hot
    const float* W1  = (const float*)d_in[2];
    const float* b1  = (const float*)d_in[3];
    const float* W2  = (const float*)d_in[4];
    const float* b2  = (const float*)d_in[5];
    const float* W3  = (const float*)d_in[6];
    const float* b3  = (const float*)d_in[7];
    const float* Ws1 = (const float*)d_in[8];
    const float* bs1 = (const float*)d_in[9];
    const float* Ws2 = (const float*)d_in[10];
    const float* bs2 = (const float*)d_in[11];
    const float* Ws3 = (const float*)d_in[12];
    const float* bs3 = (const float*)d_in[13];
    float* dout = (float*)d_out;

    // workspace carve-up (weights stored transposed [N][K] bf16)
    char* p = (char*)d_ws;
    u16* agg_bf = (u16*)p;  p += (size_t)512 * 512 * 2;
    u16* W1T    = (u16*)p;  p += (size_t)1024 * 512 * 2;
    u16* W2T    = (u16*)p;  p += (size_t)512 * 1024 * 2;
    u16* Ws1aT  = (u16*)p;  p += (size_t)1024 * 512 * 2;
    u16* Ws1bT  = (u16*)p;  p += (size_t)1024 * 512 * 2;
    u16* Ws2T   = (u16*)p;  p += (size_t)512 * 1024 * 2;
    u16* h1bf   = (u16*)p;  p += (size_t)512 * 1024 * 2;
    float* h2f  = (float*)p; p += (size_t)512 * 512 * 4;
    float* a_f  = (float*)p; p += (size_t)512 * 1024 * 4;
    float* b_f  = (float*)p; p += (size_t)512 * 1024 * 4;
    int* segy   = (int*)p;   p += 512 * 4;
    float* lossAcc  = (float*)p; p += 256;
    float* tildeAcc = (float*)p; p += 3072 * 4;

    // zero accumulators (harness does not re-poison between replays)
    k_zero<<<1, 256, 0, stream>>>(lossAcc, 2);
    k_zero<<<(3072 + 255) / 256, 256, 0, stream>>>(tildeAcc, 3072);

    // conversions: A-side plain, weights transposed
    k_cvt_bf16<<<(262144 + 255) / 256, 256, 0, stream>>>(agg, agg_bf, 262144);
    k_cvt_t_bf16<<<(524288 + 255) / 256, 256, 0, stream>>>(W1, W1T, 512, 1024);
    k_cvt_t_bf16<<<(524288 + 255) / 256, 256, 0, stream>>>(W2, W2T, 1024, 512);
    k_cvt_t_bf16<<<(524288 + 255) / 256, 256, 0, stream>>>(Ws1, Ws1aT, 512, 1024);
    k_cvt_t_bf16<<<(524288 + 255) / 256, 256, 0, stream>>>(
        Ws1 + (size_t)512 * 1024, Ws1bT, 512, 1024);
    k_cvt_t_bf16<<<(524288 + 255) / 256, 256, 0, stream>>>(Ws2, Ws2T, 1024, 512);

    const size_t gemm_smem = (size_t)64 * 40 * 2 + (size_t)128 * 40 * 2; // 15360 B

    // h1 = relu(agg @ W1 + b1)  [512,1024] bf16
    k_gemm_bf16<<<dim3(8, 8), 256, gemm_smem, stream>>>(
        agg_bf, W1T, b1, nullptr, h1bf, 512, 1024, 1);
    // h2 = relu(h1 @ W2 + b2)   [512,512] f32
    k_gemm_bf16<<<dim3(8, 4), 256, gemm_smem, stream>>>(
        h1bf, W2T, b2, h2f, nullptr, 1024, 512, 1);
    // a = agg @ Ws1[:D]         [512,1024] f32
    k_gemm_bf16<<<dim3(8, 8), 256, gemm_smem, stream>>>(
        agg_bf, Ws1aT, nullptr, a_f, nullptr, 512, 1024, 0);
    // b = agg @ Ws1[D:]         [512,1024] f32
    k_gemm_bf16<<<dim3(8, 8), 256, gemm_smem, stream>>>(
        agg_bf, Ws1bT, nullptr, b_f, nullptr, 512, 1024, 0);

    // per-segment head: hat_p, seg_y, loss1
    k_head<<<2, 256, 0, stream>>>(h2f, W3, b3, y, dout, segy, lossAcc);

    // fused pairwise: 2 batches * 256 i * 4 j-blocks = 2048 WGs
    const size_t pair_smem = (size_t)64 * 516 * 4; // 132096 B
    k_pair<<<2048, 256, pair_smem, stream>>>(
        a_f, b_f, bs1, Ws2T, bs2, Ws3, bs3, dout + 1, segy, tildeAcc, lossAcc);

    // loss + normalized tilde_p
    k_final<<<2, 256, 0, stream>>>(tildeAcc, lossAcc, dout);
}
